// OneToManyMatcher_60361470378237
// MI455X (gfx1250) — matrix operand order repaired and checked by reading, not compiled, not run on hardware
//
#include <hip/hip_runtime.h>
#include <hip/hip_bf16.h>

typedef __attribute__((ext_vector_type(16))) _Float16 v16h;
typedef __attribute__((ext_vector_type(8)))  float    v8f;

#define BB   4
#define QQ   200
#define NN   32
#define DD   65536
#define QP   208          // 13 tiles of 16 rows (padded)
#define MT   13
#define NSEG 64
#define DSEG (DD / NSEG)  // 1024 -> 32 K-steps per wave

// ws float offsets
#define OFF_INTER 0
#define OFF_PM    (BB * QP * NN)            // 26624
#define OFF_GT    (OFF_PM + BB * QP)        // 27456
#define OFF_COMB  (OFF_GT + BB * NN)        // 27584
#define ZERO_FLOATS OFF_COMB                // zero inter+pm+gt

#define NLOG2E 1.44269504088896340736f

// fast sigmoid: v_mul + v_exp_f32 + v_add + v_rcp_f32 (no IEEE div sequence)
__device__ __forceinline__ float sigf(float x) {
    const float e = __builtin_amdgcn_exp2f(x * -NLOG2E);
    return __builtin_amdgcn_rcpf(1.0f + e);
}

// ---------------------------------------------------------------------------
// Kernel 1: split-K WMMA GEMM  inter[b,q,n] = sum_d sigmoid(pred) * gt
// Row/column sums are computed on the (otherwise idle) WMMA pipe via an
// all-ones fragment instead of scalar VALU reductions.
// one wave (32 threads) per block; grid = (13 M-tiles, 4 batches, 64 K-segs)
// ---------------------------------------------------------------------------
__global__ __launch_bounds__(32)
void k1_gemm(const float* __restrict__ pred, const float* __restrict__ gt,
             float* __restrict__ ws) {
    const int lane = threadIdx.x;        // 0..31
    const int mt   = blockIdx.x;         // 0..12
    const int b    = blockIdx.y;         // 0..3
    const int sg   = blockIdx.z;         // 0..NSEG-1
    const int q0   = mt * 16;
    const int m    = lane & 15;
    const bool hi  = lane >= 16;
    const int q    = q0 + m;
    const bool qok = q < QQ;
    const bool gtb = (mt == 0);          // uniform: this block owns gt sums

    const float* rowA  = pred + (size_t)(b * QQ + (qok ? q : 0)) * DD;
    const int n0       = lane & 15;
    const float* rowB0 = gt + (size_t)(b * NN + n0) * DD;
    const float* rowB1 = gt + (size_t)(b * NN + n0 + 16) * DD;

    const int klo  = hi ? 16 : 0;   // A-fragment K layout (contiguous, matches B)
    const int koff = hi ? 16 : 0;   // B-fragment K layout (32x16)

    // all-ones f16 fragment: valid as both A(16x32)=1 and B(32x16)=1
    v16h ones;
    #pragma unroll
    for (int e = 0; e < 16; ++e) ones[e] = (_Float16)1.0f;

    v8f c0 = {};   // inter, n 0..15
    v8f c1 = {};   // inter, n 16..31
    v8f c2 = {};   // A x ones  -> pred row sums (every column identical)
    v8f c3 = {};   // ones x B0 -> gt col sums n 0..15 (every row identical)
    v8f c4 = {};   // ones x B1 -> gt col sums n 16..31
    const int dBase = sg * DSEG;

    for (int kk = 0; kk < DSEG; kk += 32) {
        const int d0 = dBase + kk;

        // ---- A fragment: sigmoid(pred) -> f16, contiguous K layout ----
        v16h a;
        if (qok) {
            const float* p = rowA + d0 + klo;
            float4 f0 = *(const float4*)(p);
            float4 f1 = *(const float4*)(p + 4);
            float4 f2 = *(const float4*)(p + 8);
            float4 f3 = *(const float4*)(p + 12);
            float s[16] = { f0.x, f0.y, f0.z, f0.w,  f1.x, f1.y, f1.z, f1.w,
                            f2.x, f2.y, f2.z, f2.w,  f3.x, f3.y, f3.z, f3.w };
            #pragma unroll
            for (int e = 0; e < 16; ++e) a[e] = (_Float16)sigf(s[e]);
        } else {
            #pragma unroll
            for (int e = 0; e < 16; ++e) a[e] = (_Float16)0.0f;
        }

        // ---- B fragments: gt columns n0 and n0+16 ----
        v16h bb0, bb1;
        {
            const float* p = rowB0 + d0 + koff;
            float4 g0 = *(const float4*)(p);
            float4 g1 = *(const float4*)(p + 4);
            float4 g2 = *(const float4*)(p + 8);
            float4 g3 = *(const float4*)(p + 12);
            float t[16] = { g0.x, g0.y, g0.z, g0.w,  g1.x, g1.y, g1.z, g1.w,
                            g2.x, g2.y, g2.z, g2.w,  g3.x, g3.y, g3.z, g3.w };
            #pragma unroll
            for (int e = 0; e < 16; ++e) bb0[e] = (_Float16)t[e];
        }
        {
            const float* p = rowB1 + d0 + koff;
            float4 g0 = *(const float4*)(p);
            float4 g1 = *(const float4*)(p + 4);
            float4 g2 = *(const float4*)(p + 8);
            float4 g3 = *(const float4*)(p + 12);
            float t[16] = { g0.x, g0.y, g0.z, g0.w,  g1.x, g1.y, g1.z, g1.w,
                            g2.x, g2.y, g2.z, g2.w,  g3.x, g3.y, g3.z, g3.w };
            #pragma unroll
            for (int e = 0; e < 16; ++e) bb1[e] = (_Float16)t[e];
        }

        // ---- matrix pipe: inter tiles + row sums (+ gt col sums, mt==0) ----
        c0 = __builtin_amdgcn_wmma_f32_16x16x32_f16(false, a, false, bb0,
                                                    (short)0, c0, false, false);
        c1 = __builtin_amdgcn_wmma_f32_16x16x32_f16(false, a, false, bb1,
                                                    (short)0, c1, false, false);
        c2 = __builtin_amdgcn_wmma_f32_16x16x32_f16(false, a, false, ones,
                                                    (short)0, c2, false, false);
        if (gtb) {  // uniform scalar branch: EXEC remains all-1s for WMMA
            c3 = __builtin_amdgcn_wmma_f32_16x16x32_f16(false, ones, false, bb0,
                                                        (short)0, c3, false, false);
            c4 = __builtin_amdgcn_wmma_f32_16x16x32_f16(false, ones, false, bb1,
                                                        (short)0, c4, false, false);
        }
    }

    float* inter = ws + OFF_INTER;
    float* pm    = ws + OFF_PM;
    float* gts   = ws + OFF_GT;

    // C/D layout: n = lane&15, m = r + (lane>=16 ? 8 : 0)
    const int nC = lane & 15;
    const int mB = hi ? 8 : 0;
    #pragma unroll
    for (int r = 0; r < 8; ++r) {
        const int qq = q0 + mB + r;   // < 208, padded region absorbs q>=200
        atomicAdd(&inter[((size_t)b * QP + qq) * NN + nC], c0[r]);
        atomicAdd(&inter[((size_t)b * QP + qq) * NN + nC + 16], c1[r]);
    }

    // pred row sums: every column of c2 is the row sum; lanes 0 and 16
    // (nC==0) cover rows 0..7 and 8..15 of the tile
    if (nC == 0) {
        #pragma unroll
        for (int r = 0; r < 8; ++r) {
            const int qq = q0 + mB + r;
            atomicAdd(&pm[b * QP + qq], c2[r]);
        }
    }

    // gt col sums: every row of c3/c4 is the column sum; use row 0 (lanes 0..15)
    if (gtb && !hi) {
        atomicAdd(&gts[b * NN + n0], c3[0]);
        atomicAdd(&gts[b * NN + n0 + 16], c4[0]);
    }
}

// ---------------------------------------------------------------------------
// Kernel 2: combined = IoU * softmax(logits)[...,1]
// ---------------------------------------------------------------------------
__global__ void k2_final(const float* __restrict__ logits, float* __restrict__ ws) {
    const int i = blockIdx.x * blockDim.x + threadIdx.x;
    if (i >= BB * QQ * NN) return;
    const int n = i & 31;
    const int q = (i >> 5) % QQ;
    const int b = i / (QQ * NN);

    const float* inter = ws + OFF_INTER;
    const float* pm    = ws + OFF_PM;
    const float* gts   = ws + OFF_GT;
    float* comb        = ws + OFF_COMB;

    const float it  = inter[((size_t)b * QP + q) * NN + n];
    const float un  = pm[b * QP + q] + gts[b * NN + n] - it;
    const float iou = it * __builtin_amdgcn_rcpf(un + 1e-6f);
    const float l0  = logits[(size_t)(b * QQ + q) * 2 + 0];
    const float l1  = logits[(size_t)(b * QQ + q) * 2 + 1];
    const float cls = sigf(l1 - l0);  // softmax(2-class)[...,1]
    comb[i] = iou * cls;
}

// ---------------------------------------------------------------------------
// Kernel 3: greedy per-GT top-4 with dedup; one wave32 per batch
// ---------------------------------------------------------------------------
__global__ __launch_bounds__(32)
void k3_greedy(const float* __restrict__ comb_all, float* __restrict__ out) {
    const int b    = blockIdx.x;
    const int lane = threadIdx.x;
    __shared__ int assigned[QQ];
    for (int qi = lane; qi < QQ; qi += 32) assigned[qi] = 0;
    __syncthreads();

    const float* comb = comb_all + (size_t)b * QQ * NN;

    for (int n = 0; n < NN; ++n) {
        float colv[7];
        #pragma unroll
        for (int j = 0; j < 7; ++j) {
            const int qi = lane + 32 * j;
            colv[j] = (qi < QQ) ? comb[(size_t)qi * NN + n] : -1e30f;
        }
        unsigned pick = 0;
        int sel[4];
        #pragma unroll 1
        for (int t = 0; t < 4; ++t) {
            float bv = -1e30f; int bi = -1;
            #pragma unroll
            for (int j = 0; j < 7; ++j) {
                const int qi = lane + 32 * j;
                const float v = ((qi < QQ) && !(pick & (1u << j))) ? colv[j] : -1e30f;
                if (v > bv) { bv = v; bi = qi; }   // ascending -> lowest q on tie
            }
            // wave32 butterfly argmax, tie -> lowest index (matches lax.top_k)
            #pragma unroll
            for (int off = 16; off >= 1; off >>= 1) {
                const float ov = __shfl_xor(bv, off, 32);
                const int   oi = __shfl_xor(bi, off, 32);
                if (ov > bv || (ov == bv && (unsigned)oi < (unsigned)bi)) {
                    bv = ov; bi = oi;
                }
            }
            sel[t] = bi;
            if (bi >= 0 && lane == (bi & 31)) pick |= 1u << (bi >> 5);
        }
        if (lane == 0) {
            #pragma unroll
            for (int t = 0; t < 4; ++t) {
                const int s = sel[t];
                int taken = 1;
                if (s >= 0) { taken = assigned[s]; assigned[s] = 1; }
                const int o = (b * NN + n) * 4 + t;
                out[o]        = taken ? -1.0f : (float)s;   // src_idx
                out[512 + o]  = taken ? -1.0f : (float)n;   // tgt_idx
                out[1024 + o] = taken ? 0.0f : 1.0f;        // valid
            }
        }
    }
}

// ---------------------------------------------------------------------------
extern "C" void kernel_launch(void* const* d_in, const int* in_sizes, int n_in,
                              void* d_out, int out_size, void* d_ws, size_t ws_size,
                              hipStream_t stream) {
    const float* pred   = (const float*)d_in[0];  // [4,200,256,256]
    const float* logits = (const float*)d_in[1];  // [4,200,2]
    const float* gt     = (const float*)d_in[2];  // [4,32,256,256]
    float* ws  = (float*)d_ws;
    float* out = (float*)d_out;

    // zero split-K accumulators (graph-capture safe)
    hipMemsetAsync(ws, 0, (size_t)ZERO_FLOATS * sizeof(float), stream);

    dim3 g1(MT, BB, NSEG);
    k1_gemm<<<g1, 32, 0, stream>>>(pred, gt, ws);

    const int total = BB * QQ * NN;
    k2_final<<<(total + 255) / 256, 256, 0, stream>>>(logits, ws);

    k3_greedy<<<BB, 32, 0, stream>>>(ws + OFF_COMB, out);
}
